// S2CNNRegressor_16458314678905
// MI455X (gfx1250) — compile-verified, matrix operand order checked by reading, not run on hardware
//
#include <hip/hip_runtime.h>
#include <math.h>
#include <stdint.h>

// ---------------------------------------------------------------------------
// S2CNN regressor forward for gfx1250 (MI455X).
//  - All per-degree spectral contractions and the MLP run on v_wmma_f32_16x16x32_bf16
//    (bf16 operands, f32 accumulate; complex GEMM = 4 WMMA / K-step).
//  - DFT + Wigner stages fused into LDS-resident kernels (no 300+MB spectra);
//    twiddles come from an LDS table (no TRANS ops in inner loops).
//  - SO3-analysis tiles staged global->LDS with GLOBAL_LOAD_ASYNC_TO_LDS_B32
//    (ASYNCcnt) -- the CDNA5 async copy path.
//  - Largest live field ~193MB ~= the 192MB L2, so stages chain through L2.
// ---------------------------------------------------------------------------

#define USE_ASYNC_LDS 1

typedef __attribute__((ext_vector_type(16))) __bf16 v16bf;
typedef __attribute__((ext_vector_type(8)))  float  v8f;

static constexpr int    NAI  = 64 * 23;           // batch*atoms = 1472 (multiple of 16)
static constexpr double PI_D = 3.14159265358979323846;
static constexpr float  PI_F = 3.14159265358979323846f;

__host__ __device__ constexpr int pref_sq(int l) { return (4 * l * l * l - l) / 3; } // sum_{i<l} (2i+1)^2
__host__ __device__ constexpr int pref_m(int l)  { return l * l; }                    // sum_{i<l} (2i+1)

__device__ __forceinline__ int iabs_(int a) { return a < 0 ? -a : a; }
__device__ __forceinline__ int imax_(int a, int b) { return a > b ? a : b; }
__device__ __forceinline__ int modp_(int a, int p) { int r = a % p; return (r < 0) ? r + p : r; }

// ----------------------------- Wigner-d (f64) ------------------------------
__device__ double factd(int n) { double r = 1.0; for (int i = 2; i <= n; i++) r *= (double)i; return r; }

__device__ double wig_d(int l, int mp, int m, double beta) {
  double cb = cos(beta * 0.5), sb = sin(beta * 0.5);
  double pref = sqrt(factd(l + mp) * factd(l - mp) * factd(l + m) * factd(l - m));
  int s0 = (m - mp > 0) ? (m - mp) : 0;
  int s1 = (l + m < l - mp) ? (l + m) : (l - mp);
  double acc = 0.0;
  for (int s = s0; s <= s1; s++) {
    double t = ((mp - m + s) & 1) ? -1.0 : 1.0;
    t /= factd(l + m - s) * factd(s) * factd(mp - m + s) * factd(l - mp - s);
    t *= pow(cb, (double)(2 * l + m - mp - 2 * s)) * pow(sb, (double)(mp - m + 2 * s));
    acc += t;
  }
  return pref * acc;
}

// --------------------------- table fill helpers -----------------------------
__device__ void fill_dfft(float* dst, int nl, int bin, const double* qwv, int t, int nt) {
  int total = 2 * bin * pref_sq(nl);
  for (int idx = t; idx < total; idx += nt) {
    int rem = idx, l = 0;
    while (rem >= 2 * bin * (2 * l + 1) * (2 * l + 1)) { rem -= 2 * bin * (2 * l + 1) * (2 * l + 1); l++; }
    int Ml = 2 * l + 1, k = rem / (Ml * Ml), r = rem % (Ml * Ml), im = r / Ml, in = r % Ml;
    double beta = PI_D * (2.0 * k + 1.0) / (4.0 * bin);
    dst[idx] = (float)(wig_d(l, im - l, in - l, beta) * qwv[k]);
  }
}
__device__ void fill_difft(float* dst, int nl, int t, int nt) {
  int P = 2 * nl, total = P * pref_sq(nl);
  for (int idx = t; idx < total; idx += nt) {
    int rem = idx, l = 0;
    while (rem >= P * (2 * l + 1) * (2 * l + 1)) { rem -= P * (2 * l + 1) * (2 * l + 1); l++; }
    int Ml = 2 * l + 1, k = rem / (Ml * Ml), r = rem % (Ml * Ml), im = r / Ml, in = r % Ml;
    double beta = PI_D * (2.0 * k + 1.0) / (4.0 * nl);
    dst[idx] = (float)(wig_d(l, im - l, in - l, beta) * (2.0 * l + 1.0));
  }
}
__device__ void fill_dgrid_so3(float* dst, int nl, int t, int nt) {
  int total = 2 * pref_sq(nl);
  for (int idx = t; idx < total; idx += nt) {
    int rem = idx, l = 0;
    while (rem >= 2 * (2 * l + 1) * (2 * l + 1)) { rem -= 2 * (2 * l + 1) * (2 * l + 1); l++; }
    int Ml = 2 * l + 1, jb = rem / (Ml * Ml), r = rem % (Ml * Ml), im = r / Ml, in = r % Ml;
    double beta = (jb + 1) * PI_D / 16.0;  // near-identity grid betas
    dst[idx] = (float)wig_d(l, im - l, in - l, beta);
  }
}
__device__ void fill_dgrid_s2(float* dst, int t, int nt) {
  int total = 2 * pref_m(8);
  for (int idx = t; idx < total; idx += nt) {
    int rem = idx, l = 0;
    while (rem >= 2 * (2 * l + 1)) { rem -= 2 * (2 * l + 1); l++; }
    int Ml = 2 * l + 1, jb = rem / Ml, im = rem % Ml;
    double beta = (jb + 1) * PI_D / 16.0;
    dst[idx] = (float)wig_d(l, im - l, 0, beta);
  }
}

__global__ void k_tables(float* qw2, float* dl0,
                         float* dfft1, float* dfft2, float* dfft3,
                         float* difft0, float* difft1, float* difft2, float* difft3,
                         float* dg0, float* dg1, float* dg2, float* dg3) {
  __shared__ double qw[5][20];
  const int bl[5] = {10, 8, 6, 4, 2};
  int t = threadIdx.x, nt = blockDim.x;
  for (int s = 0; s < 5; s++) {
    int b = bl[s];
    for (int j = t; j < 2 * b; j += nt) {
      double acc = 0.0;
      for (int k = 0; k < b; k++)
        acc += sin((2.0 * j + 1.0) * (2.0 * k + 1.0) * PI_D / (4.0 * b)) / (2.0 * k + 1.0);
      qw[s][j] = 2.0 / b * sin(PI_D * (2.0 * j + 1.0) / (4.0 * b)) * acc;
    }
  }
  __syncthreads();
  for (int j = t; j < 4; j += nt) qw2[j] = (float)qw[4][j];
  for (int idx = t; idx < 20 * 64; idx += nt) { // s2 analysis table (with qw(b=10))
    int rem = idx, l = 0;
    while (rem >= 20 * (2 * l + 1)) { rem -= 20 * (2 * l + 1); l++; }
    int Ml = 2 * l + 1, b = rem / Ml, im = rem % Ml;
    double beta = PI_D * (2.0 * b + 1.0) / 40.0;
    dl0[idx] = (float)(wig_d(l, im - l, 0, beta) * qw[0][b]);
  }
  fill_dfft(dfft1, 6, 8, qw[1], t, nt);
  fill_dfft(dfft2, 4, 6, qw[2], t, nt);
  fill_dfft(dfft3, 2, 4, qw[3], t, nt);
  fill_difft(difft0, 8, t, nt);
  fill_difft(difft1, 6, t, nt);
  fill_difft(difft2, 4, t, nt);
  fill_difft(difft3, 2, t, nt);
  fill_dgrid_s2(dg0, t, nt);
  fill_dgrid_so3(dg1, 6, t, nt);
  fill_dgrid_so3(dg2, 4, t, nt);
  fill_dgrid_so3(dg3, 2, t, nt);
}

// --------------------------- rotated kernel bases ---------------------------
__global__ void k_basis_s2(const float* __restrict__ kern, const float* __restrict__ dg0,
                           float2* __restrict__ Y0, float scale) {
  const int I = 5, O = 8, NA = 20, G = 40;
  int total = I * O * pref_m(8);
  for (int idx = blockIdx.x * blockDim.x + threadIdx.x; idx < total; idx += gridDim.x * blockDim.x) {
    int rem = idx, l = 0;
    while (rem >= I * O * (2 * l + 1)) { rem -= I * O * (2 * l + 1); l++; }
    int Ml = 2 * l + 1;
    int i = rem / (O * Ml), r = rem % (O * Ml), o = r / Ml, im = r % Ml, m = im - l;
    float ax = 0.f, ay = 0.f;
    for (int jb = 0; jb < 2; jb++) {
      float d = dg0[2 * pref_m(l) + jb * Ml + im];
      for (int ia = 0; ia < NA; ia++) {
        float ph = -(float)m * (2.f * PI_F * ia / NA);
        float kv = kern[(i * O + o) * G + jb * NA + ia];
        ax += kv * d * cosf(ph); ay += kv * d * sinf(ph);
      }
    }
    Y0[idx] = make_float2(scale * ax, scale * ay);
  }
}

__global__ void k_basis_so3(const float* __restrict__ kern, const float* __restrict__ dg,
                            float2* __restrict__ Y, int I, int O, int n_alpha, int nl, float scale) {
  int total = I * O * pref_sq(nl);
  int G = 4 * n_alpha;
  for (int idx = blockIdx.x * blockDim.x + threadIdx.x; idx < total; idx += gridDim.x * blockDim.x) {
    int rem = idx, l = 0;
    while (rem >= I * O * (2 * l + 1) * (2 * l + 1)) { rem -= I * O * (2 * l + 1) * (2 * l + 1); l++; }
    int Ml = 2 * l + 1;
    int i = rem / (O * Ml * Ml), r = rem % (O * Ml * Ml), o = r / (Ml * Ml), r2 = r % (Ml * Ml);
    int im = r2 / Ml, in = r2 % Ml, m = im - l, n = in - l;
    float ax = 0.f, ay = 0.f;
    for (int jb = 0; jb < 2; jb++) {
      float d = dg[2 * pref_sq(l) + (jb * Ml + im) * Ml + in];
      for (int ia = 0; ia < n_alpha; ia++) {
        float alpha = 2.f * PI_F * ia / n_alpha;
        for (int ig = 0; ig < 2; ig++) {
          float gamma = (ig == 0) ? -PI_F / 8.f : PI_F / 8.f;
          float ph = -((float)m * alpha + (float)n * gamma);
          float kv = kern[((size_t)i * O + o) * G + (jb * n_alpha + ia) * 2 + ig];
          ax += kv * d * cosf(ph); ay += kv * d * sinf(ph);
        }
      }
    }
    Y[idx] = make_float2(scale * ax, scale * ay);
  }
}

// ------------------------------- S2 analysis --------------------------------
// one block per (a,i): alpha-DFT (len 20, LDS twiddles) then quadrature over 20 betas.
__global__ void k_s2fft(const float* __restrict__ x, const float* __restrict__ dl0,
                        float2* __restrict__ X0) {
  const int I = 5, P = 20;
  __shared__ float  tile[P * P];
  __shared__ float2 xf[P * 15];     // [beta][m+7]
  __shared__ float2 tw[P];
  int ai = blockIdx.x;
  const float* src = x + (size_t)ai * P * P;
  for (int i = threadIdx.x; i < P * P; i += blockDim.x) tile[i] = src[i];
  for (int j = threadIdx.x; j < P; j += blockDim.x)
    tw[j] = make_float2(cosf(2.f * PI_F * j / P), sinf(2.f * PI_F * j / P));
  __syncthreads();
  for (int i = threadIdx.x; i < P * 15; i += blockDim.x) {
    int b = i / 15, mi = i % 15, m = mi - 7;
    int step = modp_(m, P), e = 0;
    float sr = 0.f, si = 0.f;
    for (int p = 0; p < P; p++) {
      float v = tile[b * P + p];
      sr += v * tw[e].x; si -= v * tw[e].y;    // e^{-i m p 2pi/P}
      e += step; if (e >= P) e -= P;
    }
    xf[i] = make_float2(sr, si);
  }
  __syncthreads();
  for (int idx = threadIdx.x; idx < 64; idx += blockDim.x) {
    int rem = idx, l = 0;
    while (rem >= 2 * l + 1) { rem -= 2 * l + 1; l++; }
    int im = rem, m = im - l, Ml = 2 * l + 1;
    float2 acc = make_float2(0.f, 0.f);
    for (int b = 0; b < P; b++) {
      float w = dl0[P * pref_m(l) + b * Ml + im];
      float2 v = xf[b * 15 + (m + 7)];
      acc.x += w * v.x; acc.y += w * v.y;
    }
    X0[(size_t)NAI * I * pref_m(l) + (size_t)ai * Ml + im] = acc;
  }
}

// ------------------------------- SO3 analysis -------------------------------
// one block per (a,i): per-beta separable 2D DFT + Wigner quadrature, accumulated in LDS.
// Tiles staged global->LDS via GLOBAL_LOAD_ASYNC_TO_LDS_B32 (ASYNCcnt path).
__global__ void k_so3fft(const float* __restrict__ h, const float* __restrict__ dfft,
                         float2* __restrict__ X, int I, int bin, int nl) {
  int P = 2 * bin, L = nl - 1, W = 2 * L + 1, nx = pref_sq(nl);
  extern __shared__ char smem[];
  float*  tile = (float*)smem;                 // P*P
  float2* trow = (float2*)(tile + P * P);      // W*P
  float2* F    = trow + W * P;                 // W*W
  float2* xacc = F + W * W;                    // nx
  float2* tw   = xacc + nx;                    // P
  int ai = blockIdx.x;
  for (int i = threadIdx.x; i < nx; i += blockDim.x) xacc[i] = make_float2(0.f, 0.f);
  for (int j = threadIdx.x; j < P; j += blockDim.x)
    tw[j] = make_float2(cosf(2.f * PI_F * j / P), sinf(2.f * PI_F * j / P));
  float inv = 1.f / (float)(P * P);
  for (int k = 0; k < P; k++) {
    __syncthreads();
    const float* src = h + ((size_t)ai * P + k) * P * P;
#if USE_ASYNC_LDS
    for (int i = threadIdx.x; i < P * P; i += blockDim.x) {
      unsigned int ldsa = (unsigned int)(uintptr_t)(tile + i);   // LDS byte offset = addr[31:0]
      unsigned long long ga = (unsigned long long)(uintptr_t)(src + i);
      asm volatile("global_load_async_to_lds_b32 %0, %1, off"
                   :: "v"(ldsa), "v"(ga) : "memory");
    }
    asm volatile("s_wait_asynccnt 0" ::: "memory");
#else
    for (int i = threadIdx.x; i < P * P; i += blockDim.x) tile[i] = src[i];
#endif
    __syncthreads();
    for (int i = threadIdx.x; i < W * P; i += blockDim.x) {
      int mi = i / P, q = i % P, m = mi - L;
      int step = modp_(m, P), e = 0;
      float sr = 0.f, si = 0.f;
      for (int p = 0; p < P; p++) {
        float v = tile[p * P + q];
        sr += v * tw[e].x; si -= v * tw[e].y;    // e^{-i m p 2pi/P}
        e += step; if (e >= P) e -= P;
      }
      trow[i] = make_float2(sr, si);
    }
    __syncthreads();
    for (int i = threadIdx.x; i < W * W; i += blockDim.x) {
      int mi = i / W, ni = i % W, n = ni - L;
      int step = modp_(n, P), e = 0;
      float2 acc = make_float2(0.f, 0.f);
      for (int q = 0; q < P; q++) {
        float c = tw[e].x, s = -tw[e].y;         // e^{-i n q 2pi/P}
        float2 v = trow[mi * P + q];
        acc.x += v.x * c - v.y * s; acc.y += v.x * s + v.y * c;
        e += step; if (e >= P) e -= P;
      }
      F[i] = make_float2(acc.x * inv, acc.y * inv);
    }
    __syncthreads();
    for (int idx = threadIdx.x; idx < nx; idx += blockDim.x) {
      int rem = idx, l = 0;
      while (rem >= (2 * l + 1) * (2 * l + 1)) { rem -= (2 * l + 1) * (2 * l + 1); l++; }
      int Ml = 2 * l + 1, im = rem / Ml, in = rem % Ml, m = im - l, n = in - l;
      float d = dfft[P * pref_sq(l) + (k * Ml + im) * Ml + in];
      float2 v = F[(m + L) * W + (n + L)];
      xacc[idx].x += d * v.x; xacc[idx].y += d * v.y;
    }
  }
  __syncthreads();
  for (int idx = threadIdx.x; idx < nx; idx += blockDim.x) {
    int rem = idx, l = 0;
    while (rem >= (2 * l + 1) * (2 * l + 1)) { rem -= (2 * l + 1) * (2 * l + 1); l++; }
    int Ml = 2 * l + 1, im = rem / Ml, in = rem % Ml;
    X[(size_t)NAI * I * pref_sq(l) + ((size_t)ai * Ml + im) * Ml + in] = xacc[idx];
  }
}

// ------------------------------ pack kernels --------------------------------
__global__ void k_pack_A_s2(const float2* __restrict__ Xl, __bf16* __restrict__ Ar,
                            __bf16* __restrict__ Ai, int Ml, int I, int Kpad) {
  long long total = (long long)NAI * Ml * Kpad;
  for (long long idx = blockIdx.x * (long long)blockDim.x + threadIdx.x; idx < total;
       idx += (long long)gridDim.x * blockDim.x) {
    int row = (int)(idx / Kpad), k = (int)(idx % Kpad);
    float re = 0.f, im = 0.f;
    if (k < I) {
      int a = row / Ml, m = row % Ml;
      float2 v = Xl[((size_t)a * I + k) * Ml + m];
      re = v.x; im = v.y;
    }
    Ar[idx] = (__bf16)re; Ai[idx] = (__bf16)im;
  }
}
__global__ void k_pack_A_so3(const float2* __restrict__ Xl, __bf16* __restrict__ Ar,
                             __bf16* __restrict__ Ai, int Ml, int I, int Kpad) {
  long long total = (long long)NAI * Ml * Kpad;
  int K = I * Ml;
  for (long long idx = blockIdx.x * (long long)blockDim.x + threadIdx.x; idx < total;
       idx += (long long)gridDim.x * blockDim.x) {
    int row = (int)(idx / Kpad), k = (int)(idx % Kpad);
    float re = 0.f, im = 0.f;
    if (k < K) {
      int a = row / Ml, m = row % Ml, i = k / Ml, kk = k % Ml;
      float2 v = Xl[(((size_t)a * I + i) * Ml + m) * Ml + kk];
      re = v.x; im = v.y;
    }
    Ar[idx] = (__bf16)re; Ai[idx] = (__bf16)im;
  }
}
__global__ void k_pack_Bt_s2(const float2* __restrict__ Yl, __bf16* __restrict__ Btr,
                             __bf16* __restrict__ Bti, int Ml, int I, int O, int Kpad, int Npad) {
  long long total = (long long)Npad * Kpad;
  for (long long idx = blockIdx.x * (long long)blockDim.x + threadIdx.x; idx < total;
       idx += (long long)gridDim.x * blockDim.x) {
    int nrow = (int)(idx / Kpad), k = (int)(idx % Kpad);
    float re = 0.f, im = 0.f;
    if (nrow < O * Ml && k < I) {
      int o = nrow / Ml, n = nrow % Ml;
      float2 v = Yl[((size_t)k * O + o) * Ml + n];
      re = v.x; im = v.y;   // conj folded into GEMM sign combination
    }
    Btr[idx] = (__bf16)re; Bti[idx] = (__bf16)im;
  }
}
__global__ void k_pack_Bt_so3(const float2* __restrict__ Yl, __bf16* __restrict__ Btr,
                              __bf16* __restrict__ Bti, int Ml, int I, int O, int Kpad, int Npad) {
  long long total = (long long)Npad * Kpad;
  int K = I * Ml;
  for (long long idx = blockIdx.x * (long long)blockDim.x + threadIdx.x; idx < total;
       idx += (long long)gridDim.x * blockDim.x) {
    int nrow = (int)(idx / Kpad), k = (int)(idx % Kpad);
    float re = 0.f, im = 0.f;
    if (nrow < O * Ml && k < K) {
      int o = nrow / Ml, n = nrow % Ml, i = k / Ml, kk = k % Ml;
      float2 v = Yl[(((size_t)i * O + o) * Ml + n) * Ml + kk];  // einsum n -> Y's m index
      re = v.x; im = v.y;
    }
    Btr[idx] = (__bf16)re; Bti[idx] = (__bf16)im;
  }
}
__global__ void k_pack_real_A(const float* __restrict__ X, __bf16* __restrict__ Ar,
                              int M, int K, int Kpad) {
  long long total = (long long)M * Kpad;
  for (long long idx = blockIdx.x * (long long)blockDim.x + threadIdx.x; idx < total;
       idx += (long long)gridDim.x * blockDim.x) {
    int row = (int)(idx / Kpad), k = (int)(idx % Kpad);
    Ar[idx] = (__bf16)((k < K) ? X[(size_t)row * K + k] : 0.f);
  }
}
__global__ void k_pack_real_Bt(const float* __restrict__ Wkn, __bf16* __restrict__ Btr,
                               int K, int N, int Kpad, int Npad) {
  long long total = (long long)Npad * Kpad;
  for (long long idx = blockIdx.x * (long long)blockDim.x + threadIdx.x; idx < total;
       idx += (long long)gridDim.x * blockDim.x) {
    int n = (int)(idx / Kpad), k = (int)(idx % Kpad);
    Btr[idx] = (__bf16)((n < N && k < K) ? Wkn[(size_t)k * N + n] : 0.f);
  }
}

// ---------------------------- WMMA GEMM kernels -----------------------------
// ISA 16-bit A/B fragment: per lane two contiguous 8-bf16 runs at k0+8*half and k0+16+8*half.
__device__ __forceinline__ v16bf load_frag(const __bf16* __restrict__ rowptr, int k0, int half) {
  const __bf16* p0 = rowptr + k0 + 8 * half;
  v16bf f;
#pragma unroll
  for (int e = 0; e < 8; e++) f[e] = p0[e];
#pragma unroll
  for (int e = 0; e < 8; e++) f[8 + e] = p0[16 + e];
  return f;
}

// Z = A * conj(B): Re = ArBr + AiBi, Im = AiBr - ArBi. One wave per 16x16 tile.
__global__ void k_cgemm_conjB(const __bf16* __restrict__ Ar, const __bf16* __restrict__ Ai,
                              const __bf16* __restrict__ Btr, const __bf16* __restrict__ Bti,
                              float2* __restrict__ Z, int M, int N, int Kpad, int Npad) {
  int ntn = Npad / 16;
  int wave = (blockIdx.x * blockDim.x + threadIdx.x) >> 5;
  int nt = (M / 16) * ntn;
  if (wave >= nt) return;                       // wave-uniform: EXEC stays all-ones
  int lane = threadIdx.x & 31, half = lane >> 4, ln = lane & 15;
  int tm = wave / ntn, tn = wave % ntn;
  const __bf16* arp = Ar  + (size_t)(tm * 16 + ln) * Kpad;
  const __bf16* aip = Ai  + (size_t)(tm * 16 + ln) * Kpad;
  const __bf16* brp = Btr + (size_t)(tn * 16 + ln) * Kpad;
  const __bf16* bip = Bti + (size_t)(tn * 16 + ln) * Kpad;
  v8f accR = {}, accI1 = {}, accI2 = {};
  for (int k0 = 0; k0 < Kpad; k0 += 32) {
    __builtin_prefetch(arp + k0 + 64, 0, 1);
    __builtin_prefetch(brp + k0 + 64, 0, 1);
    v16bf far_ = load_frag(arp, k0, half);
    v16bf fai_ = load_frag(aip, k0, half);
    v16bf fbr_ = load_frag(brp, k0, half);
    v16bf fbi_ = load_frag(bip, k0, half);
    accR  = __builtin_amdgcn_wmma_f32_16x16x32_bf16(false, far_, false, fbr_, (short)0, accR,  false, false);
    accR  = __builtin_amdgcn_wmma_f32_16x16x32_bf16(false, fai_, false, fbi_, (short)0, accR,  false, false);
    accI1 = __builtin_amdgcn_wmma_f32_16x16x32_bf16(false, fai_, false, fbr_, (short)0, accI1, false, false);
    accI2 = __builtin_amdgcn_wmma_f32_16x16x32_bf16(false, far_, false, fbi_, (short)0, accI2, false, false);
  }
#pragma unroll
  for (int v = 0; v < 8; v++) {
    int gm = tm * 16 + v + half * 8;            // C/D layout: VGPR v -> row v (+8 for upper half)
    int gn = tn * 16 + ln;
    if (gn < N) Z[(size_t)gm * N + gn] = make_float2(accR[v], accI1[v] - accI2[v]);
  }
}

__global__ void k_rgemm_wmma(const __bf16* __restrict__ A, const __bf16* __restrict__ Bt,
                             const float* __restrict__ bias, float* __restrict__ C,
                             int M, int N, int Kpad, int Npad) {
  int ntn = Npad / 16;
  int wave = (blockIdx.x * blockDim.x + threadIdx.x) >> 5;
  int nt = (M / 16) * ntn;
  if (wave >= nt) return;
  int lane = threadIdx.x & 31, half = lane >> 4, ln = lane & 15;
  int tm = wave / ntn, tn = wave % ntn;
  const __bf16* ap = A  + (size_t)(tm * 16 + ln) * Kpad;
  const __bf16* bp = Bt + (size_t)(tn * 16 + ln) * Kpad;
  v8f acc = {};
  for (int k0 = 0; k0 < Kpad; k0 += 32) {
    __builtin_prefetch(ap + k0 + 64, 0, 1);
    v16bf fa = load_frag(ap, k0, half);
    v16bf fb = load_frag(bp, k0, half);
    acc = __builtin_amdgcn_wmma_f32_16x16x32_bf16(false, fa, false, fb, (short)0, acc, false, false);
  }
#pragma unroll
  for (int v = 0; v < 8; v++) {
    int gm = tm * 16 + v + half * 8, gn = tn * 16 + ln;
    if (gn < N) C[(size_t)gm * N + gn] = acc[v] + (bias ? bias[gn] : 0.f);
  }
}

// --------------------- SO3 synthesis + bias + BN statistics ------------------
// one block per (a,o); accumulates per-channel sum/sumsq with one atomicAdd pair.
__global__ void k_ifft_bnstat(const float2* __restrict__ Z, const float* __restrict__ difft,
                              const float* __restrict__ bias, float* __restrict__ h,
                              float* __restrict__ stats, int O, int nl) {
  int P = 2 * nl, L = nl - 1, W = 2 * L + 1, nz = pref_sq(nl);
  extern __shared__ char smem[];
  float2* zc = (float2*)smem;            // nz
  float2* wk = zc + nz;                  // W*W
  float2* tt = wk + W * W;               // W*P
  float2* tw = tt + W * P;               // P
  float*  red = (float*)(tw + P);        // 2*blockDim
  int a = blockIdx.x / O, o = blockIdx.x % O;
  for (int idx = threadIdx.x; idx < nz; idx += blockDim.x) {
    int rem = idx, l = 0;
    while (rem >= (2 * l + 1) * (2 * l + 1)) { rem -= (2 * l + 1) * (2 * l + 1); l++; }
    int Ml = 2 * l + 1, im = rem / Ml, in = rem % Ml;
    zc[idx] = Z[(size_t)NAI * O * pref_sq(l) + ((size_t)a * Ml + im) * ((size_t)O * Ml) + (size_t)o * Ml + in];
  }
  for (int j = threadIdx.x; j < P; j += blockDim.x)
    tw[j] = make_float2(cosf(2.f * PI_F * j / P), sinf(2.f * PI_F * j / P));
  float lsum = 0.f, lsq = 0.f;
  float bo = bias[o];
  for (int k = 0; k < P; k++) {
    __syncthreads();
    for (int i = threadIdx.x; i < W * W; i += blockDim.x) {
      int mi = i / W, ni = i % W, m = mi - L, n = ni - L;
      float2 acc = make_float2(0.f, 0.f);
      int lmin = imax_(iabs_(m), iabs_(n));
      for (int l = lmin; l < nl; l++) {
        int Ml = 2 * l + 1;
        float d = difft[P * pref_sq(l) + (k * Ml + (m + l)) * Ml + (n + l)];
        float2 v = zc[pref_sq(l) + (m + l) * Ml + (n + l)];
        acc.x += d * v.x; acc.y += d * v.y;
      }
      wk[i] = acc;
    }
    __syncthreads();
    for (int i = threadIdx.x; i < W * P; i += blockDim.x) {
      int mi = i / P, j2 = i % P;
      int step = j2, e = modp_(-(L * j2), P);    // n from -L upward
      float2 acc = make_float2(0.f, 0.f);
      for (int ni = 0; ni < W; ni++) {
        float c = tw[e].x, s = tw[e].y;          // e^{+i n j2 2pi/P}
        float2 v = wk[mi * W + ni];
        acc.x += v.x * c - v.y * s; acc.y += v.x * s + v.y * c;
        e += step; if (e >= P) e -= P;
      }
      tt[i] = acc;
    }
    __syncthreads();
    for (int i = threadIdx.x; i < P * P; i += blockDim.x) {
      int j1 = i / P, j2 = i % P;
      int step = j1, e = modp_(-(L * j1), P);    // m from -L upward
      float acc = 0.f;
      for (int mi = 0; mi < W; mi++) {
        float2 v = tt[mi * P + j2];
        acc += v.x * tw[e].x - v.y * tw[e].y;    // Re(t * e^{+i m j1 2pi/P})
        e += step; if (e >= P) e -= P;
      }
      float yv = acc + bo;
      h[((size_t)blockIdx.x * P + k) * P * P + i] = yv;
      lsum += yv; lsq += yv * yv;
    }
  }
  __syncthreads();
  red[threadIdx.x] = lsum; red[blockDim.x + threadIdx.x] = lsq;
  __syncthreads();
  for (int s = blockDim.x >> 1; s > 0; s >>= 1) {
    if ((int)threadIdx.x < s) {
      red[threadIdx.x] += red[threadIdx.x + s];
      red[blockDim.x + threadIdx.x] += red[blockDim.x + threadIdx.x + s];
    }
    __syncthreads();
  }
  if (threadIdx.x == 0) {
    atomicAdd(stats + 2 * o, red[0]);
    atomicAdd(stats + 2 * o + 1, red[blockDim.x]);
  }
}

__global__ void k_bn_relu3(float* __restrict__ h, const float* __restrict__ stats,
                           const float* __restrict__ g, const float* __restrict__ b,
                           int O, int vol) {
  size_t total = (size_t)NAI * O * vol;
  float cnt = (float)NAI * (float)vol;
  for (size_t idx = (size_t)blockIdx.x * blockDim.x + threadIdx.x; idx < total;
       idx += (size_t)gridDim.x * blockDim.x) {
    int o = (int)((idx / vol) % O);
    float mu = stats[2 * o] / cnt;
    float var = stats[2 * o + 1] / cnt - mu * mu;
    float v = (h[idx] - mu) * rsqrtf(var + 1e-5f) * g[o] + b[o];
    h[idx] = fmaxf(v, 0.f);
  }
}

// ------------------------------ tail kernels --------------------------------
__global__ void k_integrate(const float* __restrict__ h, const float* __restrict__ qw2,
                            float* __restrict__ out) {
  int idx = blockIdx.x * blockDim.x + threadIdx.x;
  if (idx >= NAI * 64) return;
  const float* src = h + (size_t)idx * 64;   // [k=4][4][4]
  float acc = 0.f;
  for (int k = 0; k < 4; k++) {
    float s = 0.f;
    for (int j = 0; j < 16; j++) s += src[k * 16 + j];
    acc += (s * (1.f / 16.f)) * qw2[k];
  }
  out[idx] = acc;
}

__global__ void k_colstats(const float* __restrict__ X, int rows, int cols,
                           float* __restrict__ mu, float* __restrict__ var) {
  int c = blockIdx.x;
  __shared__ float rs[256], rq[256];
  float s = 0.f, q = 0.f;
  for (int r = threadIdx.x; r < rows; r += blockDim.x) {
    float v = X[(size_t)r * cols + c];
    s += v; q += v * v;
  }
  rs[threadIdx.x] = s; rq[threadIdx.x] = q;
  __syncthreads();
  for (int st = 128; st > 0; st >>= 1) {
    if ((int)threadIdx.x < st) { rs[threadIdx.x] += rs[threadIdx.x + st]; rq[threadIdx.x] += rq[threadIdx.x + st]; }
    __syncthreads();
  }
  if (threadIdx.x == 0) {
    float m = rs[0] / rows;
    mu[c] = m; var[c] = rq[0] / rows - m * m;
  }
}

__global__ void k_bn_relu_rows(float* __restrict__ X, const float* __restrict__ mu,
                               const float* __restrict__ var, const float* __restrict__ g,
                               const float* __restrict__ b, int rows, int cols) {
  size_t total = (size_t)rows * cols;
  for (size_t idx = (size_t)blockIdx.x * blockDim.x + threadIdx.x; idx < total;
       idx += (size_t)gridDim.x * blockDim.x) {
    int c = (int)(idx % cols);
    float v = (X[idx] - mu[c]) * rsqrtf(var[c] + 1e-5f) * g[c] + b[c];
    X[idx] = fmaxf(v, 0.f);
  }
}

__global__ void k_masksum(const float* __restrict__ t2, const int* __restrict__ atom,
                          float* __restrict__ hm) {
  int idx = blockIdx.x * blockDim.x + threadIdx.x;
  if (idx >= 64 * 64) return;
  int b = idx / 64, f = idx % 64;
  float s = 0.f;
  for (int n = 0; n < 23; n++)
    if (atom[b * 23 + n] > 0) s += t2[((size_t)b * 23 + n) * 64 + f];
  hm[idx] = s;
}

__global__ void k_final(const float* __restrict__ t3, const float* __restrict__ W4,
                        const float* __restrict__ b4, float* __restrict__ out) {
  int r = blockIdx.x * blockDim.x + threadIdx.x;
  if (r >= 64) return;
  float s = b4[0];
  for (int c = 0; c < 512; c++) s += t3[(size_t)r * 512 + c] * W4[c];
  out[r] = s;
}

__global__ void k_zero(float* p, int n) {
  int i = blockIdx.x * blockDim.x + threadIdx.x;
  if (i < n) p[i] = 0.f;
}

// --------------------------------- driver -----------------------------------
static inline int rnd16(int n) { return (n + 15) & ~15; }
static inline int rnd32(int n) { return (n + 31) & ~31; }
static inline int blks(long long n, int t) { return (int)((n + t - 1) / t); }

extern "C" void kernel_launch(void* const* d_in, const int* in_sizes, int n_in,
                              void* d_out, int out_size, void* d_ws, size_t ws_size,
                              hipStream_t stream) {
  (void)in_sizes; (void)n_in; (void)out_size; (void)ws_size;
  const float* x_in = (const float*)d_in[0];
  const int*   atom = (const int*)d_in[1];
  const float* kern_[4] = {(const float*)d_in[2], (const float*)d_in[6], (const float*)d_in[10], (const float*)d_in[14]};
  const float* bias_[4] = {(const float*)d_in[3], (const float*)d_in[7], (const float*)d_in[11], (const float*)d_in[15]};
  const float* bng_[4]  = {(const float*)d_in[4], (const float*)d_in[8], (const float*)d_in[12], (const float*)d_in[16]};
  const float* bnb_[4]  = {(const float*)d_in[5], (const float*)d_in[9], (const float*)d_in[13], (const float*)d_in[17]};
  const float* W1 = (const float*)d_in[18]; const float* b1 = (const float*)d_in[19];
  const float* gw1 = (const float*)d_in[20]; const float* gb1 = (const float*)d_in[21];
  const float* W2 = (const float*)d_in[22]; const float* b2 = (const float*)d_in[23];
  const float* gw2 = (const float*)d_in[24]; const float* gb2 = (const float*)d_in[25];
  const float* W3 = (const float*)d_in[26]; const float* b3 = (const float*)d_in[27];
  const float* gw3 = (const float*)d_in[28]; const float* gb3 = (const float*)d_in[29];
  const float* W4 = (const float*)d_in[30]; const float* b4 = (const float*)d_in[31];
  float* out = (float*)d_out;

  // ---- workspace carve (pointer math only; no allocation) ----
  char* p = (char*)d_ws;
  auto alloc = [&](size_t bytes) -> void* {
    char* r = p; p += (bytes + 255) & ~(size_t)255; return (void*)r;
  };
  float* qw2    = (float*)alloc(4 * 4);
  float* dl0    = (float*)alloc(20 * 64 * 4);
  float* dfft1  = (float*)alloc(16 * 286 * 4);
  float* dfft2  = (float*)alloc(12 * 84 * 4);
  float* dfft3  = (float*)alloc(8 * 10 * 4);
  float* difft0 = (float*)alloc(16 * 680 * 4);
  float* difft1 = (float*)alloc(12 * 286 * 4);
  float* difft2 = (float*)alloc(8 * 84 * 4);
  float* difft3 = (float*)alloc(4 * 10 * 4);
  float* dg0    = (float*)alloc(2 * 64 * 4);
  float* dg1    = (float*)alloc(2 * 286 * 4);
  float* dg2    = (float*)alloc(2 * 84 * 4);
  float* dg3    = (float*)alloc(2 * 10 * 4);
  float2* Y[4];
  Y[0] = (float2*)alloc((size_t)40 * 64 * 8);
  Y[1] = (float2*)alloc((size_t)128 * 286 * 8);
  Y[2] = (float2*)alloc((size_t)512 * 84 * 8);
  Y[3] = (float2*)alloc((size_t)2048 * 10 * 8);
  float2* X[4];
  X[0] = (float2*)alloc((size_t)NAI * 5 * 64 * 8);
  X[1] = (float2*)alloc((size_t)NAI * 8 * 286 * 8);
  X[2] = (float2*)alloc((size_t)NAI * 16 * 84 * 8);
  X[3] = (float2*)alloc((size_t)NAI * 32 * 10 * 8);
  float2* Zb = (float2*)alloc((size_t)NAI * 8 * 680 * 8);     // max over stages (64MB)
  float*  hb = (float*)alloc((size_t)NAI * 8 * 4096 * 4);     // max field (~193MB ~ L2)
  __bf16* Ar = (__bf16*)alloc((size_t)1600000 * 2);
  __bf16* Ai = (__bf16*)alloc((size_t)1600000 * 2);
  __bf16* Br = (__bf16*)alloc((size_t)40000 * 2);
  __bf16* Bi = (__bf16*)alloc((size_t)40000 * 2);
  float* stats = (float*)alloc(2048 * 4);
  float* muB   = (float*)alloc(512 * 4);
  float* varB  = (float*)alloc(512 * 4);
  float* gint  = (float*)alloc((size_t)NAI * 64 * 4);
  float* t1    = (float*)alloc((size_t)NAI * 256 * 4);
  float* t2    = (float*)alloc((size_t)NAI * 64 * 4);
  float* hm    = (float*)alloc(64 * 64 * 4);
  float* t3    = (float*)alloc(64 * 512 * 4);

  // ---- tables + rotated-kernel bases ----
  k_tables<<<1, 256, 0, stream>>>(qw2, dl0, dfft1, dfft2, dfft3,
                                  difft0, difft1, difft2, difft3, dg0, dg1, dg2, dg3);
  k_basis_s2<<<10, 256, 0, stream>>>(kern_[0], dg0, Y[0], 1.f / sqrtf(40.f * 5.f));
  k_basis_so3<<<256, 256, 0, stream>>>(kern_[1], dg1, Y[1], 8, 16, 16, 6, 1.f / sqrtf(64.f * 8.f));
  k_basis_so3<<<256, 256, 0, stream>>>(kern_[2], dg2, Y[2], 16, 32, 12, 4, 1.f / sqrtf(48.f * 16.f));
  k_basis_so3<<<128, 256, 0, stream>>>(kern_[3], dg3, Y[3], 32, 64, 8, 2, 1.f / sqrtf(32.f * 32.f));

  const int Is[4]  = {5, 8, 16, 32};
  const int Os[4]  = {8, 16, 32, 64};
  const int bins[4] = {10, 8, 6, 4};
  const int nls[4] = {8, 6, 4, 2};
  float* dffts[4]  = {nullptr, dfft1, dfft2, dfft3};
  float* diffts[4] = {difft0, difft1, difft2, difft3};

  // ---- stage 0: S2 block ----
  k_s2fft<<<NAI * 5, 256, 0, stream>>>(x_in, dl0, X[0]);
  for (int l = 0; l < 8; l++) {
    int Ml = 2 * l + 1, M = NAI * Ml, Kpad = 32, N = 8 * Ml, Npad = rnd16(N);
    k_pack_A_s2<<<blks((long long)M * Kpad, 256), 256, 0, stream>>>(
        X[0] + (size_t)NAI * 5 * pref_m(l), Ar, Ai, Ml, 5, Kpad);
    k_pack_Bt_s2<<<blks((long long)Npad * Kpad, 256), 256, 0, stream>>>(
        Y[0] + (size_t)40 * pref_m(l), Br, Bi, Ml, 5, 8, Kpad, Npad);
    int tiles = (M / 16) * (Npad / 16);
    k_cgemm_conjB<<<(tiles + 3) / 4, 128, 0, stream>>>(
        Ar, Ai, Br, Bi, Zb + (size_t)NAI * 8 * pref_sq(l), M, N, Kpad, Npad);
  }
  {
    int O = 8, nl = 8, P = 2 * nl, W = 2 * nl - 1, vol = P * P * P;
    k_zero<<<1, 64, 0, stream>>>(stats, 2 * O);
    size_t lds = (size_t)pref_sq(nl) * 8 + (size_t)W * W * 8 + (size_t)W * P * 8 +
                 (size_t)P * 8 + 2 * 256 * 4;
    k_ifft_bnstat<<<NAI * O, 256, lds, stream>>>(Zb, difft0, bias_[0], hb, stats, O, nl);
    k_bn_relu3<<<blks((long long)NAI * O * vol, 256), 256, 0, stream>>>(hb, stats, bng_[0], bnb_[0], O, vol);
  }

  // ---- stages 1..3: SO3 blocks ----
  for (int s = 1; s <= 3; s++) {
    int I = Is[s], O = Os[s], bin = bins[s], nl = nls[s];
    int Pin = 2 * bin, Win = 2 * nl - 1;
    size_t lds_f = (size_t)Pin * Pin * 4 + (size_t)Win * Pin * 8 + (size_t)Win * Win * 8 +
                   (size_t)pref_sq(nl) * 8 + (size_t)Pin * 8;
    k_so3fft<<<NAI * I, 256, lds_f, stream>>>(hb, dffts[s], X[s], I, bin, nl);
    for (int l = 0; l < nl; l++) {
      int Ml = 2 * l + 1, M = NAI * Ml, K = I * Ml, Kpad = rnd32(K), N = O * Ml, Npad = rnd16(N);
      k_pack_A_so3<<<blks((long long)M * Kpad, 256), 256, 0, stream>>>(
          X[s] + (size_t)NAI * I * pref_sq(l), Ar, Ai, Ml, I, Kpad);
      k_pack_Bt_so3<<<blks((long long)Npad * Kpad, 256), 256, 0, stream>>>(
          Y[s] + (size_t)I * O * pref_sq(l), Br, Bi, Ml, I, O, Kpad, Npad);
      int tiles = (M / 16) * (Npad / 16);
      k_cgemm_conjB<<<(tiles + 3) / 4, 128, 0, stream>>>(
          Ar, Ai, Br, Bi, Zb + (size_t)NAI * O * pref_sq(l), M, N, Kpad, Npad);
    }
    int Pout = 2 * nl, vol = Pout * Pout * Pout;
    k_zero<<<1, 128, 0, stream>>>(stats, 2 * O);
    size_t lds_i = (size_t)pref_sq(nl) * 8 + (size_t)Win * Win * 8 + (size_t)Win * Pout * 8 +
                   (size_t)Pout * 8 + 2 * 256 * 4;
    k_ifft_bnstat<<<NAI * O, 256, lds_i, stream>>>(Zb, diffts[s], bias_[s], hb, stats, O, nl);
    k_bn_relu3<<<blks((long long)NAI * O * vol, 256), 256, 0, stream>>>(hb, stats, bng_[s], bnb_[s], O, vol);
  }

  // ---- SO3 integrate + MLP head (WMMA real GEMMs) ----
  k_integrate<<<blks((long long)NAI * 64, 256), 256, 0, stream>>>(hb, qw2, gint);

  // layer 1: (1472x64) @ (64x256)
  k_pack_real_A<<<blks((long long)NAI * 64, 256), 256, 0, stream>>>(gint, Ar, NAI, 64, 64);
  k_pack_real_Bt<<<blks(256LL * 64, 256), 256, 0, stream>>>(W1, Br, 64, 256, 64, 256);
  k_rgemm_wmma<<<((NAI / 16) * (256 / 16) + 3) / 4, 128, 0, stream>>>(Ar, Br, b1, t1, NAI, 256, 64, 256);
  k_colstats<<<256, 256, 0, stream>>>(t1, NAI, 256, muB, varB);
  k_bn_relu_rows<<<blks((long long)NAI * 256, 256), 256, 0, stream>>>(t1, muB, varB, gw1, gb1, NAI, 256);

  // layer 2: (1472x256) @ (256x64)
  k_pack_real_A<<<blks((long long)NAI * 256, 256), 256, 0, stream>>>(t1, Ar, NAI, 256, 256);
  k_pack_real_Bt<<<blks(64LL * 256, 256), 256, 0, stream>>>(W2, Br, 256, 64, 256, 64);
  k_rgemm_wmma<<<((NAI / 16) * (64 / 16) + 3) / 4, 128, 0, stream>>>(Ar, Br, b2, t2, NAI, 64, 256, 64);
  k_colstats<<<64, 256, 0, stream>>>(t2, NAI, 64, muB, varB);
  k_bn_relu_rows<<<blks((long long)NAI * 64, 256), 256, 0, stream>>>(t2, muB, varB, gw2, gb2, NAI, 64);

  // masked sum over atoms -> (64 x 64)
  k_masksum<<<blks(64LL * 64, 256), 256, 0, stream>>>(t2, atom, hm);

  // layer 3: (64x64) @ (64x512)
  k_pack_real_A<<<blks(64LL * 64, 256), 256, 0, stream>>>(hm, Ar, 64, 64, 64);
  k_pack_real_Bt<<<blks(512LL * 64, 256), 256, 0, stream>>>(W3, Br, 64, 512, 64, 512);
  k_rgemm_wmma<<<((64 / 16) * (512 / 16) + 3) / 4, 128, 0, stream>>>(Ar, Br, b3, t3, 64, 512, 64, 512);
  k_colstats<<<512, 256, 0, stream>>>(t3, 64, 512, muB, varB);
  k_bn_relu_rows<<<blks(64LL * 512, 256), 256, 0, stream>>>(t3, muB, varB, gw3, gb3, 64, 512);

  // layer 4: (64x512) @ (512x1)
  k_final<<<1, 64, 0, stream>>>(t3, W4, b4, out);
}
